// MultiHeadSparseAttention_38328288150100
// MI455X (gfx1250) — matrix operand order repaired and checked by reading, not compiled, not run on hardware
//
#include <hip/hip_runtime.h>

// Problem constants (B,N,C,H,D) = (4,1024,768,12,64)
#define BB 4
#define NN 1024
#define CC 768
#define HH 12
#define DD 64
#define MROWS (NN * BB)     // 4096 rows, row m = seq*B + b
#define NTILES_C (CC / 16)  // 48
#define MTILES (MROWS / 16) // 256
#define ATT_SCALE 0.125f    // D^-0.5

typedef __attribute__((ext_vector_type(16))) __bf16        v16bf;
typedef __attribute__((ext_vector_type(8)))  float         v8f;
typedef __attribute__((ext_vector_type(4)))  float         v4f;
typedef __attribute__((ext_vector_type(4)))  unsigned int  v4u;
typedef __attribute__((ext_vector_type(8)))  int           v8i;
typedef __attribute__((ext_vector_type(4)))  int           v4i;

union Frag {
    v16bf          v;
    unsigned short us[16];
    v4u            b128[2];
};
union H8 {
    unsigned short us[8];
    v4u            u4;
};

#define WMMA_BF16(A, Bm, Cm) \
    __builtin_amdgcn_wmma_f32_16x16x32_bf16(false, (A), false, (Bm), (short)0, (Cm), false, false)

__device__ __forceinline__ unsigned short f2bf(float f) {
    union { float f; unsigned int u; } c; c.f = f;
    unsigned int u = c.u;
    return (unsigned short)((u + 0x7FFFu + ((u >> 16) & 1u)) >> 16); // RNE
}
__device__ __forceinline__ void pack4(unsigned short* d, v4f x) {
    d[0] = f2bf(x.x); d[1] = f2bf(x.y); d[2] = f2bf(x.z); d[3] = f2bf(x.w);
}

// ---------------------------------------------------------------------------
// TDM: 2D tile load global -> LDS.  D# built per CDNA5 ISA.
// data_size = 1 (2-byte elems). Groups 2/3 zero (<=2D tensor). Issued by one
// wave only (caller guarantees a *scalar* guard), tracked with TENSORcnt.
// ---------------------------------------------------------------------------
__device__ __forceinline__ void tdm_load_tile_2d(
    const unsigned short* gsrc, unsigned lds_byte_off,
    unsigned tensor_d0, unsigned tensor_d1,
    unsigned stride_d0, unsigned tile_d0, unsigned tile_d1)
{
#if __has_builtin(__builtin_amdgcn_tensor_load_to_lds)
    unsigned long long ga = (unsigned long long)gsrc;
    v4u g0;
    g0[0] = 1u;                                   // count=1 (valid), user mode
    g0[1] = lds_byte_off;                         // lds_addr
    g0[2] = (unsigned)(ga & 0xffffffffu);         // global_addr[31:0]
    g0[3] = (unsigned)((ga >> 32) & 0x1ffffffu)   // global_addr[56:32]
          | (2u << 30);                           // type=2 ("image")
    v8i g1;
    g1[0] = (int)(1u << 16);                      // data_size=1 (2B); no pad/iter
    g1[1] = (int)((tensor_d0 & 0xffffu) << 16);   // tensor_dim0[15:0] @ bits63:48
    g1[2] = (int)((tensor_d0 >> 16) | ((tensor_d1 & 0xffffu) << 16));
    g1[3] = (int)((tensor_d1 >> 16) | (tile_d0 << 16));
    g1[4] = (int)(tile_d1 & 0xffffu);             // tile_dim1; tile_dim2=0
    g1[5] = (int)stride_d0;                       // tensor_dim0_stride[31:0]
    g1[6] = 0;                                    // stride hi=0; dim1_stride=0
    g1[7] = 0;
    v4i z4 = {0, 0, 0, 0};
#if defined(__clang_major__) && (__clang_major__ >= 23)
    v8i z8 = {0, 0, 0, 0, 0, 0, 0, 0};
    __builtin_amdgcn_tensor_load_to_lds(g0, g1, z4, z4, z8, 0);
#else
    __builtin_amdgcn_tensor_load_to_lds(g0, g1, z4, z4, 0);
#endif
#else // fallback: manual wave copy (32 lanes x 16B chunks)
    unsigned short* dst = (unsigned short*)(size_t)lds_byte_off; // not reachable path
    const unsigned lane = threadIdx.x & 31;
    const unsigned chunks = (tile_d0 * tile_d1) / 8;
    for (unsigned i = lane; i < chunks; i += 32) {
        unsigned row = (i * 8) / tile_d0, col = (i * 8) % tile_d0;
        *(v4u*)(dst + i * 8) = *(const v4u*)(gsrc + row * stride_d0 + col);
    }
#endif
}

// ---------------------------------------------------------------------------
// Kernel 0: elementwise f32 -> bf16 conversion (one pass; weights reused 256x,
// activations 48x, so converting once removes all VALU cvt from GEMM loops).
// ---------------------------------------------------------------------------
__global__ __launch_bounds__(256) void cvt_bf16_kernel(
    const float* __restrict__ src, unsigned short* __restrict__ dst)
{
    const size_t i = ((size_t)blockIdx.x * 256 + threadIdx.x) * 8;
    H8 o;
    pack4(o.us + 0, *(const v4f*)(src + i));
    pack4(o.us + 4, *(const v4f*)(src + i + 4));
    *(v4u*)(dst + i) = o.u4;
}

// ---------------------------------------------------------------------------
// Kernel 1: fused QKV projection (all-bf16 operands), epilogue bias+zeta,
// scatter: t=0 Qh[B,H,N,D], t=1 Kh[B,H,N,D], t=2 Vt[B,H,D,N].
// ---------------------------------------------------------------------------
__global__ __launch_bounds__(256) void proj_qkv_kernel(
    const unsigned short* __restrict__ xq, const unsigned short* __restrict__ xk,
    const unsigned short* __restrict__ xv,
    const unsigned short* __restrict__ wq, const unsigned short* __restrict__ wk,
    const unsigned short* __restrict__ wv,
    const float* __restrict__ bq, const float* __restrict__ bk,
    const float* __restrict__ bv, const float* __restrict__ zeta,
    unsigned short* __restrict__ Qh, unsigned short* __restrict__ Kh,
    unsigned short* __restrict__ Vt)
{
    const int lane = threadIdx.x & 31;
    const int tile = blockIdx.x * 8 + __builtin_amdgcn_readfirstlane(threadIdx.x >> 5);
    const int t    = tile / (MTILES * NTILES_C);
    const int rem  = tile % (MTILES * NTILES_C);
    const int m0   = (rem / NTILES_C) * 16;
    const int n0   = (rem % NTILES_C) * 16;

    const unsigned short* X    = (t == 0) ? xq : (t == 1) ? xk : xv;
    const unsigned short* W    = (t == 0) ? wq : (t == 1) ? wk : wv;
    const float*          bias = (t == 0) ? bq : (t == 1) ? bk : bv;

    const int l15  = lane & 15;
    const int aoff = (lane < 16) ? 0 : 16;  // A frag: low lanes K 0-15, high 16-31
    const int boff = (lane < 16) ? 0 : 16;  // B frag: low lanes K 0-15, high 16-31

    const unsigned short* Arow = X + (size_t)(m0 + l15) * CC;
    const int             ncol = n0 + l15;
    const unsigned short* Brow = W + (size_t)ncol * CC; // B(k,n)=W[n][k]

    v8f c = {0.f, 0.f, 0.f, 0.f, 0.f, 0.f, 0.f, 0.f};
    for (int k0 = 0; k0 < CC; k0 += 32) {
        Frag a, b;
        a.b128[0] = *(const v4u*)(Arow + k0 + aoff);
        a.b128[1] = *(const v4u*)(Arow + k0 + aoff + 8);
        b.b128[0] = *(const v4u*)(Brow + k0 + boff);
        b.b128[1] = *(const v4u*)(Brow + k0 + boff + 8);
        c = WMMA_BF16(a.v, b.v, c);
    }

    const float bz = bias[ncol];
    const float zc = zeta[ncol];
    const int   h = ncol >> 6, d = ncol & 63;
    const int   mhi = (lane < 16) ? 0 : 8;
#pragma unroll
    for (int r = 0; r < 8; ++r) {
        const int m   = m0 + r + mhi;
        const int seq = m >> 2;  // m = seq*B + b, B=4
        const int bb  = m & 3;
        const unsigned short val = f2bf((c[r] + bz) * zc);
        if (t == 2)      Vt[((bb * HH + h) * DD + d) * NN + seq] = val;
        else if (t == 0) Qh[((bb * HH + h) * NN + seq) * DD + d] = val;
        else             Kh[((bb * HH + h) * NN + seq) * DD + d] = val;
    }
}

// ---------------------------------------------------------------------------
// Kernel 2: flash attention. grid = B*H*8 blocks, 8 waves/block, wave = 16 q rows.
// Per 32-wide k-tile: wave0 TDM-loads K(32x64) and V^T(64x32) tiles into LDS
// (shared by all 8 waves), then 4 score WMMAs + online softmax + LDS P-relayout
// + 4 PV WMMAs per wave.
// ---------------------------------------------------------------------------
__global__ __launch_bounds__(256) void attn_kernel(
    const unsigned short* __restrict__ Qh, const unsigned short* __restrict__ Kh,
    const unsigned short* __restrict__ Vt, const unsigned char* __restrict__ mask,
    unsigned short* __restrict__ ctx)
{
    __shared__ unsigned short Klds[32 * 64];     // 4 KB: [kpos][d]
    __shared__ unsigned short Vlds[64 * 32];     // 4 KB: [d][kpos]
    __shared__ unsigned short Plds[8 * 16 * 32]; // 8 KB: per-wave P slab

    const int lane = threadIdx.x & 31;
    const int wave = __builtin_amdgcn_readfirstlane(threadIdx.x >> 5);
    const int bh   = blockIdx.x >> 3; // 0..47
    const int qblk = blockIdx.x & 7;
    const int bb   = bh / HH;
    const int h    = bh % HH;
    const int q0   = qblk * 128 + wave * 16;

    const unsigned short* Qb = Qh + (size_t)bh * NN * DD;
    const unsigned short* Kb = Kh + (size_t)bh * NN * DD;
    const unsigned short* Vb = Vt + (size_t)bh * DD * NN;
    const unsigned char*  mb = mask + bb * NN; // jax bool = 1 byte

    const unsigned kldsOff = (unsigned)(unsigned long long)(void*)Klds;
    const unsigned vldsOff = (unsigned)(unsigned long long)(void*)Vlds;

    const int l15  = lane & 15;
    const int aoff = (lane < 16) ? 0 : 16;
    const int boff = (lane < 16) ? 0 : 16;

    // Loop-invariant Q A-fragments (d-chunks 0..31 / 32..63)
    Frag aq0, aq1;
    {
        const unsigned short* row = Qb + (q0 + l15) * DD;
        aq0.b128[0] = *(const v4u*)(row + 0  + aoff);
        aq0.b128[1] = *(const v4u*)(row + 8  + aoff);
        aq1.b128[0] = *(const v4u*)(row + 32 + aoff);
        aq1.b128[1] = *(const v4u*)(row + 40 + aoff);
    }

    v8f acc0 = {0,0,0,0,0,0,0,0}, acc1 = acc0, acc2 = acc0, acc3 = acc0;
    float mrow[8], lrow[8];
#pragma unroll
    for (int r = 0; r < 8; ++r) { mrow[r] = -3.0e38f; lrow[r] = 0.f; }

    unsigned short* Pw = Plds + wave * (16 * 32);

    for (int kt = 0; kt < NN; kt += 32) {
        __syncthreads(); // previous tile's LDS readers done before DMA overwrite
        if (wave == 0) { // scalar guard: TDM ignores EXEC, must branch around
            tdm_load_tile_2d(Kb + (size_t)kt * DD, kldsOff, DD, NN, DD, DD, 32);
            tdm_load_tile_2d(Vb + kt, vldsOff, NN, DD, NN, 32, DD);
            __builtin_amdgcn_s_wait_tensorcnt(0);
        }
        __syncthreads(); // publish K/V tiles to all waves

        v8f s[2];
#pragma unroll
        for (int half = 0; half < 2; ++half) {
            Frag bk0, bk1;
            const unsigned short* krow = Klds + (16 * half + l15) * DD;
            bk0.b128[0] = *(const v4u*)(krow + boff);
            bk0.b128[1] = *(const v4u*)(krow + boff + 8);
            bk1.b128[0] = *(const v4u*)(krow + 32 + boff);
            bk1.b128[1] = *(const v4u*)(krow + 32 + boff + 8);
            v8f cs = {0,0,0,0,0,0,0,0};
            cs = WMMA_BF16(aq0.v, bk0.v, cs);
            cs = WMMA_BF16(aq1.v, bk1.v, cs);
            const float madd = mb[kt + 16 * half + l15] ? -1.0e30f : 0.0f;
#pragma unroll
            for (int r = 0; r < 8; ++r) cs[r] = (cs[r] + madd) * ATT_SCALE;
            s[half] = cs;
        }

        // Online softmax over 32 cols; row = (VGPR idx r, lane half)
#pragma unroll
        for (int r = 0; r < 8; ++r) {
            float mx = fmaxf(s[0][r], s[1][r]);
            mx = fmaxf(mx, __shfl_xor(mx, 1, 32));
            mx = fmaxf(mx, __shfl_xor(mx, 2, 32));
            mx = fmaxf(mx, __shfl_xor(mx, 4, 32));
            mx = fmaxf(mx, __shfl_xor(mx, 8, 32));
            const float mnew = fmaxf(mrow[r], mx);
            const float corr = __expf(mrow[r] - mnew);
            const float p0 = __expf(s[0][r] - mnew);
            const float p1 = __expf(s[1][r] - mnew);
            float sum = p0 + p1;
            sum += __shfl_xor(sum, 1, 32);
            sum += __shfl_xor(sum, 2, 32);
            sum += __shfl_xor(sum, 4, 32);
            sum += __shfl_xor(sum, 8, 32);
            lrow[r] = lrow[r] * corr + sum;
            mrow[r] = mnew;
            acc0[r] *= corr; acc1[r] *= corr; acc2[r] *= corr; acc3[r] *= corr;
            const int mr = r + ((lane < 16) ? 0 : 8);
            Pw[mr * 32 + l15]      = f2bf(p0);
            Pw[mr * 32 + 16 + l15] = f2bf(p1);
        }
        asm volatile("s_wait_dscnt 0" ::: "memory"); // intra-wave P publish

        Frag pa; // P as 16x32 A-fragment
        pa.b128[0] = *(const v4u*)(Pw + l15 * 32 + aoff);
        pa.b128[1] = *(const v4u*)(Pw + l15 * 32 + 8 + aoff);

        Frag bv0, bv1, bv2, bv3; // V^T from LDS: row d, contiguous k
        {
            const unsigned short* c0 = Vlds + (0  + l15) * 32 + boff;
            const unsigned short* c1 = Vlds + (16 + l15) * 32 + boff;
            const unsigned short* c2 = Vlds + (32 + l15) * 32 + boff;
            const unsigned short* c3 = Vlds + (48 + l15) * 32 + boff;
            bv0.b128[0] = *(const v4u*)(c0); bv0.b128[1] = *(const v4u*)(c0 + 8);
            bv1.b128[0] = *(const v4u*)(c1); bv1.b128[1] = *(const v4u*)(c1 + 8);
            bv2.b128[0] = *(const v4u*)(c2); bv2.b128[1] = *(const v4u*)(c2 + 8);
            bv3.b128[0] = *(const v4u*)(c3); bv3.b128[1] = *(const v4u*)(c3 + 8);
        }
        acc0 = WMMA_BF16(pa.v, bv0.v, acc0);
        acc1 = WMMA_BF16(pa.v, bv1.v, acc1);
        acc2 = WMMA_BF16(pa.v, bv2.v, acc2);
        acc3 = WMMA_BF16(pa.v, bv3.v, acc3);
        asm volatile("s_wait_dscnt 0" ::: "memory"); // WAR on P slab
    }

    // Epilogue: normalize, store ctx [M=seq*B+b, C] bf16
#pragma unroll
    for (int r = 0; r < 8; ++r) {
        const float inv = 1.0f / lrow[r];
        const int seq = q0 + r + ((lane < 16) ? 0 : 8);
        const int m   = seq * BB + bb;
        unsigned short* crow = ctx + (size_t)m * CC + h * DD;
        crow[0  + l15] = f2bf(acc0[r] * inv);
        crow[16 + l15] = f2bf(acc1[r] * inv);
        crow[32 + l15] = f2bf(acc2[r] * inv);
        crow[48 + l15] = f2bf(acc3[r] * inv);
    }
}

// ---------------------------------------------------------------------------
// Kernel 3: output projection  out = ctx @ Wo^T + bo, f32, layout [N,B,C]
// ---------------------------------------------------------------------------
__global__ __launch_bounds__(256) void proj_out_kernel(
    const unsigned short* __restrict__ ctx, const unsigned short* __restrict__ wo,
    const float* __restrict__ bo, float* __restrict__ out)
{
    const int lane = threadIdx.x & 31;
    const int tile = blockIdx.x * 8 + __builtin_amdgcn_readfirstlane(threadIdx.x >> 5);
    const int m0   = (tile / NTILES_C) * 16;
    const int n0   = (tile % NTILES_C) * 16;

    const int l15  = lane & 15;
    const int aoff = (lane < 16) ? 0 : 16;
    const int boff = (lane < 16) ? 0 : 16;

    const unsigned short* Arow = ctx + (size_t)(m0 + l15) * CC;
    const int             ncol = n0 + l15;
    const unsigned short* Brow = wo + (size_t)ncol * CC;

    v8f c = {0.f, 0.f, 0.f, 0.f, 0.f, 0.f, 0.f, 0.f};
    for (int k0 = 0; k0 < CC; k0 += 32) {
        Frag a, b;
        a.b128[0] = *(const v4u*)(Arow + k0 + aoff);
        a.b128[1] = *(const v4u*)(Arow + k0 + aoff + 8);
        b.b128[0] = *(const v4u*)(Brow + k0 + boff);
        b.b128[1] = *(const v4u*)(Brow + k0 + boff + 8);
        c = WMMA_BF16(a.v, b.v, c);
    }

    const float bz  = bo[ncol];
    const int   mhi = (lane < 16) ? 0 : 8;
#pragma unroll
    for (int r = 0; r < 8; ++r) {
        out[(size_t)(m0 + r + mhi) * CC + ncol] = c[r] + bz;
    }
}

// ---------------------------------------------------------------------------
extern "C" void kernel_launch(void* const* d_in, const int* in_sizes, int n_in,
                              void* d_out, int out_size, void* d_ws, size_t ws_size,
                              hipStream_t stream) {
    const float*         q    = (const float*)d_in[0];
    const float*         k    = (const float*)d_in[1];
    const float*         v    = (const float*)d_in[2];
    const unsigned char* mask = (const unsigned char*)d_in[3];
    const float*         Wq   = (const float*)d_in[4];
    const float*         bq   = (const float*)d_in[5];
    const float*         Wk   = (const float*)d_in[6];
    const float*         bk   = (const float*)d_in[7];
    const float*         Wv   = (const float*)d_in[8];
    const float*         bv   = (const float*)d_in[9];
    const float*         Wo   = (const float*)d_in[10];
    const float*         bo   = (const float*)d_in[11];
    const float*         zeta = (const float*)d_in[12];

    const size_t PLANE  = (size_t)MROWS * CC; // 3,145,728 elems
    const size_t WPLANE = (size_t)CC * CC;    //   589,824 elems
    unsigned short* ws  = (unsigned short*)d_ws;
    unsigned short* xq  = ws;                       // [M,C] bf16
    unsigned short* xk  = xq + PLANE;
    unsigned short* xv  = xk + PLANE;
    unsigned short* wqb = xv + PLANE;               // [C,C] bf16 x4
    unsigned short* wkb = wqb + WPLANE;
    unsigned short* wvb = wkb + WPLANE;
    unsigned short* wob = wvb + WPLANE;
    unsigned short* Qh  = wob + WPLANE;             // [B,H,N,D] bf16
    unsigned short* Kh  = Qh + PLANE;               // [B,H,N,D] bf16
    unsigned short* Vt  = Kh + PLANE;               // [B,H,D,N] bf16
    unsigned short* ctx = Vt + PLANE;               // [M,C] bf16
    // total ws: 7*PLANE + 4*WPLANE bf16 elems ~= 46.5 MB

    // One-shot f32 -> bf16 conversions (2048 elems per block)
    cvt_bf16_kernel<<<(int)(PLANE / 2048), 256, 0, stream>>>(q, xq);
    cvt_bf16_kernel<<<(int)(PLANE / 2048), 256, 0, stream>>>(k, xk);
    cvt_bf16_kernel<<<(int)(PLANE / 2048), 256, 0, stream>>>(v, xv);
    cvt_bf16_kernel<<<(int)(WPLANE / 2048), 256, 0, stream>>>(Wq, wqb);
    cvt_bf16_kernel<<<(int)(WPLANE / 2048), 256, 0, stream>>>(Wk, wkb);
    cvt_bf16_kernel<<<(int)(WPLANE / 2048), 256, 0, stream>>>(Wv, wvb);
    cvt_bf16_kernel<<<(int)(WPLANE / 2048), 256, 0, stream>>>(Wo, wob);

    proj_qkv_kernel<<<(3 * MTILES * NTILES_C) / 8, 256, 0, stream>>>(
        xq, xk, xv, wqb, wkb, wvb, bq, bk, bv, zeta, Qh, Kh, Vt);

    attn_kernel<<<BB * HH * (NN / 128), 256, 0, stream>>>(Qh, Kh, Vt, mask, ctx);

    proj_out_kernel<<<(MTILES * NTILES_C) / 8, 256, 0, stream>>>(ctx, wob, bo, (float*)d_out);
}
